// SequenceRepresentationXLSTM_73547019976764
// MI455X (gfx1250) — matrix operand order repaired and checked by reading, not compiled, not run on hardware
//
#include <hip/hip_runtime.h>
#include <math.h>
#include <stdint.h>

// ---------------------------------------------------------------------------
// Problem constants
// ---------------------------------------------------------------------------
constexpr int kH = 256, kF = 8, kT = 252, kB = 64;
constexpr int kTB = kT * kB;     // 16128 rows (t*B+b)
constexpr int kFH = kF * kH;     // 2048
constexpr long long kBTH = (long long)kB * kT * kH;

// ---------------------------------------------------------------------------
// WMMA helpers (CDNA5 wave32, v_wmma_f32_16x16x32_bf16)
// ---------------------------------------------------------------------------
typedef __attribute__((ext_vector_type(16))) __bf16 v16bf;
typedef __attribute__((ext_vector_type(8)))  float  v8f;

union FragU { v16bf v; uint4 q[2]; };

__device__ inline unsigned short f2bf(float x) {
  unsigned u = __float_as_uint(x);
  u += 0x7fffu + ((u >> 16) & 1u);          // round-to-nearest-even
  return (unsigned short)(u >> 16);
}
__device__ inline float bf2f(unsigned short b) { return __uint_as_float(((unsigned)b) << 16); }
__device__ inline float sigmoidf_(float x) { return 1.f / (1.f + expf(-x)); }
__device__ inline float eluf_(float x) { return x > 0.f ? x : (expf(x) - 1.f); }
__device__ inline float geluf_(float x) { return 0.5f * x * (1.f + erff(x * 0.70710678118654752f)); }

// Async global->LDS staging (gfx1250 ASYNCcnt path). Fallback: plain 16B copy.
#define USE_ASYNC_LDS 1
__device__ inline void cp16_g2l(void* ldst, const void* gsrc) {
#if USE_ASYNC_LDS
  unsigned loff = (unsigned)(size_t)ldst;   // low 32 bits of generic shared ptr = LDS offset
  asm volatile("global_load_async_to_lds_b128 %0, %1, off" :: "v"(loff), "v"(gsrc) : "memory");
#else
  *(uint4*)ldst = *(const uint4*)gsrc;
#endif
}
__device__ inline void cp_join() {
#if USE_ASYNC_LDS
  asm volatile("s_wait_asynccnt 0" ::: "memory");
#endif
  __syncthreads();
}

__device__ inline v8f wmma_bf(const FragU& a, const FragU& b, v8f c) {
  return __builtin_amdgcn_wmma_f32_16x16x32_bf16(false, a.v, false, b.v, (short)0, c, false, false);
}

// A fragment 16x32 bf16 from row-major buffer (stride in halves).
// lane L: M=L&15, K = k0 + 8*(L>>4) + {0..7, 16..23}  -> two b128 loads (matches ISA table)
__device__ inline void load_a(FragU& f, const unsigned short* A, int stride, int k0) {
  int lane = threadIdx.x & 31;
  const unsigned short* p = A + (size_t)(lane & 15) * stride + k0 + 8 * (lane >> 4);
  f.q[0] = *(const uint4*)p;
  f.q[1] = *(const uint4*)(p + 16);
}
// B fragment 32x16 bf16 from transposed weights WT[N][K] (row-major, K halves per row).
// lane L: N=n0+(L&15), K = k0 + 8*(L>>4) + {0..7, 16..23} -> two b128 loads
__device__ inline void load_b(FragU& f, const unsigned short* WT, int K, int n0, int k0) {
  int lane = threadIdx.x & 31;
  const unsigned short* p = WT + (size_t)(n0 + (lane & 15)) * K + k0 + 8 * (lane >> 4);
  f.q[0] = *(const uint4*)p;
  f.q[1] = *(const uint4*)(p + 16);
}

// ---------------------------------------------------------------------------
// K0: transpose+convert weights W[K][N] f32 -> WT[N][K] bf16 (batched)
// ---------------------------------------------------------------------------
__global__ void k_wconv(const float* __restrict__ W, unsigned short* __restrict__ WT,
                        int K, int N, int batch) {
  size_t gid = (size_t)blockIdx.x * blockDim.x + threadIdx.x;
  size_t tot = (size_t)batch * K * N;
  if (gid >= tot) return;
  size_t kn = (size_t)K * N;
  size_t b = gid / kn, r = gid % kn;
  size_t n = r / K, k = r % K;
  WT[b * kn + n * K + k] = f2bf(W[b * kn + k * N + n]);
}

// ---------------------------------------------------------------------------
// K1e: emb_bf16[row][f*256+h] = bf16(x[b,t,f]*pw[f,h]+pb[f,h]),  row = t*B+b
// ---------------------------------------------------------------------------
__global__ void k_embed(const float* __restrict__ seq, const float* __restrict__ pw,
                        const float* __restrict__ pb, unsigned short* __restrict__ emb) {
  size_t gid = (size_t)blockIdx.x * blockDim.x + threadIdx.x;
  if (gid >= (size_t)kTB * kFH) return;
  int row = (int)(gid >> 11);
  int rem = (int)(gid & 2047);
  int f = rem >> 8, h = rem & 255;
  int t = row >> 6, b = row & 63;
  float x = seq[((size_t)b * kT + t) * kF + f];
  emb[gid] = f2bf(x * pw[f * kH + h] + pb[f * kH + h]);
}

// ---------------------------------------------------------------------------
// K2a: tick gather
// ---------------------------------------------------------------------------
__global__ void k_gather(const float* __restrict__ table, const int* __restrict__ idx,
                         float* __restrict__ tf, unsigned short* __restrict__ tb) {
  int gid = blockIdx.x * blockDim.x + threadIdx.x;
  if (gid >= kB * kH) return;
  int b = gid >> 8, h = gid & 255;
  float v = table[(size_t)idx[b] * kH + h];
  tf[gid] = v; tb[gid] = f2bf(v);
}

// ---------------------------------------------------------------------------
// Generic fused GRN (in=hidden=out=256): out = LN(GLU(W_g*(W2*elu(W1*x+b1+ctx)+b2)+bg) + x)
// 512 threads = 16 waves; each wave owns one 16-col tile per stage (two in stage 3).
// ---------------------------------------------------------------------------
__global__ __launch_bounds__(512) void k_grn256(
    const unsigned short* __restrict__ A,   // [M][256] bf16 rows
    const float* __restrict__ R,            // [M][256] f32 residual
    const unsigned short* __restrict__ w1T, const float* __restrict__ b1,
    const float* __restrict__ h1x, int h1x_mod,  // optional per-(row%mod) context add
    const unsigned short* __restrict__ w2T, const float* __restrict__ b2,
    const unsigned short* __restrict__ wgT, const float* __restrict__ bg,
    const float* __restrict__ gam, const float* __restrict__ bet,
    float* __restrict__ outF, unsigned short* __restrict__ outB, int swizzle) {
  int row0 = blockIdx.x * 16;
  int tid = threadIdx.x, lane = tid & 31, wv = tid >> 5;
  __shared__ unsigned short s0[16 * 256];
  __shared__ unsigned short s1[16 * 256];
  __shared__ float red1[16][16], red2[16][16];
  __shared__ float mv[16][2];

  { // stage A rows (16 x 512B) via async copies
    int r = tid >> 5, c = (tid & 31) * 8;
    cp16_g2l(&s0[r * 256 + c], A + (size_t)(row0 + r) * 256 + c);
  }
  cp_join();

  int ln = lane & 15, hi = lane >> 4;
  { // stage1: s1 = bf16(elu(s0@W1 + b1 [+ctx]))
    v8f acc = {};
    for (int k0 = 0; k0 < 256; k0 += 32) {
      FragU a, b; load_a(a, s0, 256, k0); load_b(b, w1T, 256, wv * 16, k0);
      acc = wmma_bf(a, b, acc);
    }
    for (int r = 0; r < 8; r++) {
      int rr = 8 * hi + r, col = wv * 16 + ln, row = row0 + rr;
      float v = acc[r] + b1[col];
      if (h1x) v += h1x[(size_t)(row % h1x_mod) * kH + col];
      s1[rr * 256 + col] = f2bf(eluf_(v));
    }
  }
  __syncthreads();
  { // stage2: s0 = bf16(s1@W2 + b2)
    v8f acc = {};
    for (int k0 = 0; k0 < 256; k0 += 32) {
      FragU a, b; load_a(a, s1, 256, k0); load_b(b, w2T, 256, wv * 16, k0);
      acc = wmma_bf(a, b, acc);
    }
    for (int r = 0; r < 8; r++)
      s0[(8 * hi + r) * 256 + wv * 16 + ln] = f2bf(acc[r] + b2[wv * 16 + ln]);
  }
  __syncthreads();
  // stage3: gy = s0@Wg (N=512); wave wv computes a-cols [16wv) and g-cols [256+16wv)
  float yv[8];
  {
    v8f aa = {}, gg = {};
    for (int k0 = 0; k0 < 256; k0 += 32) {
      FragU a, ba, bgf; load_a(a, s0, 256, k0);
      load_b(ba, wgT, 256, wv * 16, k0);
      load_b(bgf, wgT, 256, 256 + wv * 16, k0);
      aa = wmma_bf(a, ba, aa); gg = wmma_bf(a, bgf, gg);
    }
    for (int r = 0; r < 8; r++) {
      int rr = 8 * hi + r, col = wv * 16 + ln, row = row0 + rr;
      float y = (aa[r] + bg[col]) * sigmoidf_(gg[r] + bg[256 + col]);
      yv[r] = y + R[(size_t)row * kH + col];
    }
    for (int r = 0; r < 8; r++) {
      float s = yv[r], q = yv[r] * yv[r];
      for (int m = 1; m < 16; m <<= 1) { s += __shfl_xor(s, m, 32); q += __shfl_xor(q, m, 32); }
      if (ln == 0) { red1[wv][8 * hi + r] = s; red2[wv][8 * hi + r] = q; }
    }
  }
  __syncthreads();
  if (tid < 16) {
    float s = 0.f, q = 0.f;
    for (int w = 0; w < 16; w++) { s += red1[w][tid]; q += red2[w][tid]; }
    float mu = s * (1.f / 256.f);
    float var = q * (1.f / 256.f) - mu * mu;
    mv[tid][0] = mu; mv[tid][1] = 1.f / sqrtf(var + 1e-5f);
  }
  __syncthreads();
  for (int r = 0; r < 8; r++) {
    int rr = 8 * hi + r, col = wv * 16 + ln, row = row0 + rr;
    float o = (yv[r] - mv[rr][0]) * mv[rr][1] * gam[col] + bet[col];
    size_t oi = swizzle ? ((size_t)(row & 63) * kT + (row >> 6)) * kH + col
                        : (size_t)row * kH + col;
    outF[oi] = o;
    if (outB) outB[(size_t)row * kH + col] = f2bf(o);
  }
}

// ---------------------------------------------------------------------------
// K1: per-variable GRNs. block = (row_block, f); residual emb recomputed from x.
// ---------------------------------------------------------------------------
__global__ __launch_bounds__(512) void k_vargrn(
    const float* __restrict__ seq, const float* __restrict__ pw, const float* __restrict__ pb,
    const unsigned short* __restrict__ emb,                 // [TB][2048] bf16
    const unsigned short* __restrict__ w1T, const float* __restrict__ vb1,
    const unsigned short* __restrict__ w2T, const float* __restrict__ vb2,
    const unsigned short* __restrict__ wgT, const float* __restrict__ vbg,
    const float* __restrict__ vg, const float* __restrict__ vbe,
    unsigned short* __restrict__ varB) {                    // [TB][2048] bf16
  int row0 = blockIdx.x * 16;
  int f = blockIdx.y;
  int tid = threadIdx.x, lane = tid & 31, wv = tid >> 5;
  __shared__ unsigned short s0[16 * 256];
  __shared__ unsigned short s1[16 * 256];
  __shared__ float xv[16];
  __shared__ float red1[16][16], red2[16][16];
  __shared__ float mv[16][2];

  { // A0 = emb rows for this f
    int r = tid >> 5, c = (tid & 31) * 8;
    cp16_g2l(&s0[r * 256 + c], emb + (size_t)(row0 + r) * kFH + f * kH + c);
  }
  if (tid < 16) {
    int row = row0 + tid, t = row >> 6, b = row & 63;
    xv[tid] = seq[((size_t)b * kT + t) * kF + f];
  }
  cp_join();

  const unsigned short* W1 = w1T + (size_t)f * kH * kH;
  const unsigned short* W2 = w2T + (size_t)f * kH * kH;
  const unsigned short* Wg = wgT + (size_t)f * (2 * kH) * kH;
  const float* b1 = vb1 + f * kH;
  const float* b2 = vb2 + f * kH;
  const float* bg = vbg + f * 2 * kH;
  int ln = lane & 15, hi = lane >> 4;

  { // h1 = elu(A0@W1+b1)
    v8f acc = {};
    for (int k0 = 0; k0 < 256; k0 += 32) {
      FragU a, b; load_a(a, s0, 256, k0); load_b(b, W1, 256, wv * 16, k0);
      acc = wmma_bf(a, b, acc);
    }
    for (int r = 0; r < 8; r++)
      s1[(8 * hi + r) * 256 + wv * 16 + ln] = f2bf(eluf_(acc[r] + b1[wv * 16 + ln]));
  }
  __syncthreads();
  { // h2 = h1@W2+b2
    v8f acc = {};
    for (int k0 = 0; k0 < 256; k0 += 32) {
      FragU a, b; load_a(a, s1, 256, k0); load_b(b, W2, 256, wv * 16, k0);
      acc = wmma_bf(a, b, acc);
    }
    for (int r = 0; r < 8; r++)
      s0[(8 * hi + r) * 256 + wv * 16 + ln] = f2bf(acc[r] + b2[wv * 16 + ln]);
  }
  __syncthreads();
  float yv[8];
  { // GLU + residual(emb)
    v8f aa = {}, gg = {};
    for (int k0 = 0; k0 < 256; k0 += 32) {
      FragU a, ba, bgf; load_a(a, s0, 256, k0);
      load_b(ba, Wg, 256, wv * 16, k0);
      load_b(bgf, Wg, 256, 256 + wv * 16, k0);
      aa = wmma_bf(a, ba, aa); gg = wmma_bf(a, bgf, gg);
    }
    for (int r = 0; r < 8; r++) {
      int rr = 8 * hi + r, col = wv * 16 + ln;
      float y = (aa[r] + bg[col]) * sigmoidf_(gg[r] + bg[256 + col]);
      float embv = xv[rr] * pw[f * kH + col] + pb[f * kH + col];
      yv[r] = y + embv;
    }
    for (int r = 0; r < 8; r++) {
      float s = yv[r], q = yv[r] * yv[r];
      for (int m = 1; m < 16; m <<= 1) { s += __shfl_xor(s, m, 32); q += __shfl_xor(q, m, 32); }
      if (ln == 0) { red1[wv][8 * hi + r] = s; red2[wv][8 * hi + r] = q; }
    }
  }
  __syncthreads();
  if (tid < 16) {
    float s = 0.f, q = 0.f;
    for (int w = 0; w < 16; w++) { s += red1[w][tid]; q += red2[w][tid]; }
    float mu = s * (1.f / 256.f);
    float var = q * (1.f / 256.f) - mu * mu;
    mv[tid][0] = mu; mv[tid][1] = 1.f / sqrtf(var + 1e-5f);
  }
  __syncthreads();
  for (int r = 0; r < 8; r++) {
    int rr = 8 * hi + r, col = wv * 16 + ln, row = row0 + rr;
    float o = (yv[r] - mv[rr][0]) * mv[rr][1] * vg[f * kH + col] + vbe[f * kH + col];
    varB[(size_t)row * kFH + f * kH + col] = f2bf(o);
  }
}

// ---------------------------------------------------------------------------
// K3: flattened GRN (2048 -> 8) + softmax -> selection weights (scalar path)
// ---------------------------------------------------------------------------
__global__ void k_flat(const unsigned short* __restrict__ emb, const float* __restrict__ ctxvs,
                       const float* __restrict__ w1, const float* __restrict__ b1,
                       const float* __restrict__ wc,
                       const float* __restrict__ w2, const float* __restrict__ b2,
                       const float* __restrict__ wg, const float* __restrict__ bgp,
                       const float* __restrict__ gm, const float* __restrict__ be,
                       const float* __restrict__ rw, const float* __restrict__ rb,
                       float* __restrict__ wsel) {
  int row = blockIdx.x * blockDim.x + threadIdx.x;
  if (row >= kTB) return;
  int b = row & (kB - 1);
  float h[8], r8[8];
  for (int j = 0; j < 8; j++) { h[j] = b1[j]; r8[j] = rb[j]; }
  const unsigned short* e = emb + (size_t)row * kFH;
  for (int i = 0; i < kFH; i++) {
    float x = bf2f(e[i]);
    const float* a = w1 + i * 8; const float* c = rw + i * 8;
    for (int j = 0; j < 8; j++) { h[j] += x * a[j]; r8[j] += x * c[j]; }
  }
  for (int i = 0; i < kH; i++) {
    float x = ctxvs[(size_t)b * kH + i];
    const float* a = wc + i * 8;
    for (int j = 0; j < 8; j++) h[j] += x * a[j];
  }
  for (int j = 0; j < 8; j++) h[j] = eluf_(h[j]);
  float h2[8];
  for (int j = 0; j < 8; j++) { float s = b2[j]; for (int i = 0; i < 8; i++) s += h[i] * w2[i * 8 + j]; h2[j] = s; }
  float gy[16];
  for (int j = 0; j < 16; j++) { float s = bgp[j]; for (int i = 0; i < 8; i++) s += h2[i] * wg[i * 16 + j]; gy[j] = s; }
  float y[8]; float mu = 0.f;
  for (int j = 0; j < 8; j++) { y[j] = gy[j] * sigmoidf_(gy[8 + j]) + r8[j]; mu += y[j]; }
  mu *= 0.125f;
  float var = 0.f;
  for (int j = 0; j < 8; j++) { float d = y[j] - mu; var += d * d; }
  var *= 0.125f;
  float inv = 1.f / sqrtf(var + 1e-5f);
  float lnv[8]; float mx = -1e30f;
  for (int j = 0; j < 8; j++) { lnv[j] = (y[j] - mu) * inv * gm[j] + be[j]; mx = fmaxf(mx, lnv[j]); }
  float s = 0.f;
  for (int j = 0; j < 8; j++) { lnv[j] = expf(lnv[j] - mx); s += lnv[j]; }
  float r = 1.f / s;
  for (int j = 0; j < 8; j++) wsel[(size_t)row * 8 + j] = lnv[j] * r;
}

// K3b: vsn_out[row][h] = sum_f var_out[row][f][h] * w[row][f]
__global__ void k_combine(const unsigned short* __restrict__ varB, const float* __restrict__ wsel,
                          float* __restrict__ vsnF, unsigned short* __restrict__ vsnB) {
  size_t gid = (size_t)blockIdx.x * blockDim.x + threadIdx.x;
  if (gid >= (size_t)kTB * kH) return;
  int row = (int)(gid >> 8), h = (int)(gid & 255);
  const float* w = wsel + (size_t)row * 8;
  const unsigned short* v = varB + (size_t)row * kFH + h;
  float s = 0.f;
  for (int f = 0; f < 8; f++) s += bf2f(v[f * kH]) * w[f];
  vsnF[gid] = s; vsnB[gid] = f2bf(s);
}

// ctxc[b][j] = ctx_enr[b] @ fin_wc  (f32 scalar, tiny)
__global__ void k_ctxmm(const float* __restrict__ ctx, const float* __restrict__ wc,
                        float* __restrict__ out) {
  int gid = blockIdx.x * blockDim.x + threadIdx.x;
  if (gid >= kB * kH) return;
  int b = gid >> 8, j = gid & 255;
  float s = 0.f;
  for (int h = 0; h < kH; h++) s += ctx[(size_t)b * kH + h] * wc[(size_t)h * kH + j];
  out[gid] = s;
}

// ---------------------------------------------------------------------------
// K4: xLSTM scan — persistent 64-WG kernel, atomic grid barrier, WMMA tiles.
// ---------------------------------------------------------------------------
__device__ inline void gridsync(unsigned* bar, unsigned nwg, unsigned* phase) {
  __syncthreads();
  if (threadIdx.x == 0) {
    __threadfence();
    unsigned target = *phase + nwg;
    __hip_atomic_fetch_add(bar, 1u, __ATOMIC_ACQ_REL, __HIP_MEMORY_SCOPE_AGENT);
    while (__hip_atomic_load(bar, __ATOMIC_ACQUIRE, __HIP_MEMORY_SCOPE_AGENT) < target)
      __builtin_amdgcn_s_sleep(2);
    *phase = target;
  }
  __syncthreads();
}

__device__ inline void blk_ln_stats(float v, float* rb8a, float* rb8b, float* out2) {
  float s = v, q = v * v;
  for (int m = 1; m < 32; m <<= 1) { s += __shfl_xor(s, m, 32); q += __shfl_xor(q, m, 32); }
  int lane = threadIdx.x & 31, wv = threadIdx.x >> 5;
  if (lane == 0) { rb8a[wv] = s; rb8b[wv] = q; }
  __syncthreads();
  if (threadIdx.x == 0) {
    float S = 0.f, Q = 0.f;
    for (int i = 0; i < 8; i++) { S += rb8a[i]; Q += rb8b[i]; }
    float mu = S * (1.f / 256.f);
    float var = Q * (1.f / 256.f) - mu * mu;
    out2[0] = mu; out2[1] = 1.f / sqrtf(var + 1e-5f);
  }
  __syncthreads();
}

__global__ __launch_bounds__(256) void k_xlstm(
    const unsigned short* __restrict__ vsnB,     // [TB][256] bf16
    const unsigned short* __restrict__ wgatesT,  // [768][512] (wi|wf|wo cols)
    const float* __restrict__ bi, const float* __restrict__ bfp, const float* __restrict__ bo,
    const unsigned short* __restrict__ wciT, const float* __restrict__ bci,   // [512][512]
    const unsigned short* __restrict__ wcpT, const float* __restrict__ bcp,   // [256][512]
    const float* __restrict__ lncg, const float* __restrict__ lncb,
    const float* __restrict__ lnhg, const float* __restrict__ lnhb,
    float* __restrict__ cF, unsigned short* __restrict__ hB,
    float* __restrict__ gates, unsigned short* __restrict__ interB, float* __restrict__ craw,
    float* __restrict__ hsF, unsigned short* __restrict__ hsB, unsigned* bar) {
  int tid = threadIdx.x, lane = tid & 31, wv = tid >> 5;
  int gw = blockIdx.x * 8 + wv;   // 0..511 global wave id
  __shared__ unsigned phase;
  __shared__ float rb8a[8], rb8b[8], stat[2];
  if (tid == 0) phase = 0;
  __syncthreads();
  int ln = lane & 15, hi = lane >> 4;

  for (int t = 0; t < kT; t++) {
    // --- stage 1: gates_pre[64,768] and inter_pre[64,512] from comb=[vsn_t | h]
    if (gw < 320) {
      int mt = gw / 80, nt = gw % 80;
      int n0 = nt * 16, m0 = mt * 16;
      bool isGate = n0 < 768;
      const unsigned short* WT = isGate ? wgatesT : wciT;
      int nq = isGate ? n0 : (n0 - 768);
      int m = m0 + ln;
      v8f acc = {};
      for (int k0 = 0; k0 < 512; k0 += 32) {
        FragU a, b;
        int ka = k0 + 8 * hi;
        const unsigned short* pa = (k0 < 256)
            ? (vsnB + ((size_t)(t * kB + m)) * 256 + ka)
            : (hB + (size_t)m * 256 + (ka - 256));
        a.q[0] = *(const uint4*)pa; a.q[1] = *(const uint4*)(pa + 16);
        load_b(b, WT, 512, nq, k0);
        acc = wmma_bf(a, b, acc);
      }
      for (int r = 0; r < 8; r++) {
        int mm = m0 + 8 * hi + r;
        if (isGate) {
          int col = n0 + ln;
          float bias = col < 256 ? bi[col] : (col < 512 ? bfp[col - 256] : bo[col - 512]);
          gates[(size_t)mm * 768 + col] = sigmoidf_(acc[r] + bias);
        } else {
          int col = nq + ln;
          interB[(size_t)mm * 512 + col] = f2bf(geluf_(acc[r] + bci[col]));
        }
      }
    }
    gridsync(bar, gridDim.x, &phase);
    // --- stage 2: craw = f*c + i*(inter@wcp + bcp)
    if (gw < 64) {
      int mt = gw / 16, nt = gw % 16;
      int n0 = nt * 16, m0 = mt * 16;
      v8f acc = {};
      for (int k0 = 0; k0 < 512; k0 += 32) {
        FragU a, b;
        const unsigned short* pa = interB + (size_t)(m0 + ln) * 512 + k0 + 8 * hi;
        a.q[0] = *(const uint4*)pa; a.q[1] = *(const uint4*)(pa + 16);
        load_b(b, wcpT, 512, n0, k0);
        acc = wmma_bf(a, b, acc);
      }
      for (int r = 0; r < 8; r++) {
        int mm = m0 + 8 * hi + r, col = n0 + ln;
        float ct = acc[r] + bcp[col];
        float iv = gates[(size_t)mm * 768 + col];
        float fv = gates[(size_t)mm * 768 + 256 + col];
        craw[(size_t)mm * 256 + col] = fv * cF[(size_t)mm * 256 + col] + iv * ct;
      }
    }
    gridsync(bar, gridDim.x, &phase);
    // --- stage 3: per-row LayerNorms; block r handles batch row r
    {
      int rrow = blockIdx.x, c = tid;
      float x = craw[(size_t)rrow * 256 + c];
      blk_ln_stats(x, rb8a, rb8b, stat);
      float cn = (x - stat[0]) * stat[1] * lncg[c] + lncb[c];
      cF[(size_t)rrow * 256 + c] = cn;
      float ov = gates[(size_t)rrow * 768 + 512 + c];
      float hp = ov * tanhf(cn);
      blk_ln_stats(hp, rb8a, rb8b, stat);
      float hv = (hp - stat[0]) * stat[1] * lnhg[c] + lnhb[c];
      size_t oi = (size_t)(t * kB + rrow) * 256 + c;
      hsF[oi] = hv; hsB[oi] = f2bf(hv);
      hB[(size_t)rrow * 256 + c] = f2bf(hv);
    }
    gridsync(bar, gridDim.x, &phase);
  }
}

// ---------------------------------------------------------------------------
// K5: GateAddNorm: hidden = LN(GLU(hs@gan_wg+bg) + vsn); also writes d_out[1] swizzled
// ---------------------------------------------------------------------------
__global__ __launch_bounds__(512) void k_gan(
    const unsigned short* __restrict__ A, const float* __restrict__ skip,
    const unsigned short* __restrict__ wgT, const float* __restrict__ bg,
    const float* __restrict__ gam, const float* __restrict__ bet,
    float* __restrict__ hidF, unsigned short* __restrict__ hidB, float* __restrict__ out1) {
  int row0 = blockIdx.x * 16;
  int tid = threadIdx.x, lane = tid & 31, wv = tid >> 5;
  __shared__ unsigned short s0[16 * 256];
  __shared__ float red1[16][16], red2[16][16];
  __shared__ float mv[16][2];
  {
    int r = tid >> 5, c = (tid & 31) * 8;
    cp16_g2l(&s0[r * 256 + c], A + (size_t)(row0 + r) * 256 + c);
  }
  cp_join();
  int ln = lane & 15, hi = lane >> 4;
  float yv[8];
  {
    v8f aa = {}, gg = {};
    for (int k0 = 0; k0 < 256; k0 += 32) {
      FragU a, ba, bgf; load_a(a, s0, 256, k0);
      load_b(ba, wgT, 256, wv * 16, k0);
      load_b(bgf, wgT, 256, 256 + wv * 16, k0);
      aa = wmma_bf(a, ba, aa); gg = wmma_bf(a, bgf, gg);
    }
    for (int r = 0; r < 8; r++) {
      int rr = 8 * hi + r, col = wv * 16 + ln, row = row0 + rr;
      float y = (aa[r] + bg[col]) * sigmoidf_(gg[r] + bg[256 + col]);
      yv[r] = y + skip[(size_t)row * kH + col];
    }
    for (int r = 0; r < 8; r++) {
      float s = yv[r], q = yv[r] * yv[r];
      for (int m = 1; m < 16; m <<= 1) { s += __shfl_xor(s, m, 32); q += __shfl_xor(q, m, 32); }
      if (ln == 0) { red1[wv][8 * hi + r] = s; red2[wv][8 * hi + r] = q; }
    }
  }
  __syncthreads();
  if (tid < 16) {
    float s = 0.f, q = 0.f;
    for (int w = 0; w < 16; w++) { s += red1[w][tid]; q += red2[w][tid]; }
    float mu = s * (1.f / 256.f);
    float var = q * (1.f / 256.f) - mu * mu;
    mv[tid][0] = mu; mv[tid][1] = 1.f / sqrtf(var + 1e-5f);
  }
  __syncthreads();
  for (int r = 0; r < 8; r++) {
    int rr = 8 * hi + r, col = wv * 16 + ln, row = row0 + rr;
    float o = (yv[r] - mv[rr][0]) * mv[rr][1] * gam[col] + bet[col];
    hidF[(size_t)row * kH + col] = o;
    hidB[(size_t)row * kH + col] = f2bf(o);
    out1[((size_t)(row & 63) * kT + (row >> 6)) * kH + col] = o;
  }
}

// ---------------------------------------------------------------------------
// Host side
// ---------------------------------------------------------------------------
extern "C" void kernel_launch(void* const* d_in, const int* in_sizes, int n_in,
                              void* d_out, int out_size, void* d_ws, size_t ws_size,
                              hipStream_t stream) {
  (void)in_sizes; (void)n_in; (void)out_size;
  // Input order: setup_inputs() insertion order (sequence, static_tickers, params depth-first).
  const float* seq = (const float*)d_in[0];
  const int* tickers = (const int*)d_in[1];
  auto F32 = [&](int i) { return (const float*)d_in[i]; };
  const float* pw = F32(2);  const float* pb = F32(3);
  const float* v_w1 = F32(4); const float* v_b1 = F32(5);
  const float* v_w2 = F32(6); const float* v_b2 = F32(7);
  const float* v_wg = F32(8); const float* v_bg = F32(9);
  const float* v_g = F32(10); const float* v_be = F32(11);
  const float* fl_w1 = F32(12); const float* fl_b1 = F32(13); const float* fl_wc = F32(14);
  const float* fl_w2 = F32(15); const float* fl_b2 = F32(16);
  const float* fl_wg = F32(17); const float* fl_bg = F32(18);
  const float* fl_g = F32(19);  const float* fl_be = F32(20);
  const float* fl_rw = F32(21); const float* fl_rb = F32(22);
  // sc GRNs: vs(23), enr(31), h(39), c(47): each w1,b1,w2,b2,wg,bg,g,be
  const float* ticker_emb = F32(55);
  const float* wi = F32(56); const float* bi = F32(57);
  const float* wf = F32(58); const float* bff = F32(59);
  const float* wo = F32(60); const float* bo = F32(61);
  const float* wci = F32(62); const float* bci = F32(63);
  const float* wcp = F32(64); const float* bcp = F32(65);
  const float* lnc_g = F32(66); const float* lnc_b = F32(67);
  const float* lnh_g = F32(68); const float* lnh_b = F32(69);
  const float* gan_wg = F32(70); const float* gan_bg = F32(71);
  const float* gan_g = F32(72);  const float* gan_b = F32(73);
  const float* fin_w1 = F32(74); const float* fin_b1 = F32(75); const float* fin_wc = F32(76);
  const float* fin_w2 = F32(77); const float* fin_b2 = F32(78);
  const float* fin_wg = F32(79); const float* fin_bg = F32(80);
  const float* fin_g = F32(81);  const float* fin_be = F32(82);

  // Workspace carve-out
  char* ws = (char*)d_ws;
  size_t off = 0;
  auto alloc = [&](size_t bytes) { char* p = ws + off; off += (bytes + 255) & ~(size_t)255; return p; };
  unsigned short* WT_VW1 = (unsigned short*)alloc((size_t)8 * 65536 * 2);
  unsigned short* WT_VW2 = (unsigned short*)alloc((size_t)8 * 65536 * 2);
  unsigned short* WT_VWG = (unsigned short*)alloc((size_t)8 * 131072 * 2);
  unsigned short* WT_SC  = (unsigned short*)alloc((size_t)4 * 262144 * 2);  // per g: w1T|w2T|wgT
  unsigned short* WT_GATES = (unsigned short*)alloc((size_t)768 * 512 * 2);
  unsigned short* WT_WCI = (unsigned short*)alloc((size_t)512 * 512 * 2);
  unsigned short* WT_WCP = (unsigned short*)alloc((size_t)256 * 512 * 2);
  unsigned short* WT_GAN = (unsigned short*)alloc((size_t)512 * 256 * 2);
  unsigned short* WT_FIN_W1 = (unsigned short*)alloc((size_t)65536 * 2);
  unsigned short* WT_FIN_W2 = (unsigned short*)alloc((size_t)65536 * 2);
  unsigned short* WT_FIN_WG = (unsigned short*)alloc((size_t)131072 * 2);
  unsigned short* EMB_BF = (unsigned short*)alloc((size_t)kTB * kFH * 2);
  unsigned short* VAR_BF = (unsigned short*)alloc((size_t)kTB * kFH * 2);
  float* TICK_F = (float*)alloc((size_t)kB * kH * 4);
  unsigned short* TICK_BF = (unsigned short*)alloc((size_t)kB * kH * 2);
  float* CTXVS_F = (float*)alloc((size_t)kB * kH * 4);
  float* CTXENR_F = (float*)alloc((size_t)kB * kH * 4);
  float* H0_F = (float*)alloc((size_t)kB * kH * 4);
  float* C_F = (float*)alloc((size_t)kB * kH * 4);
  unsigned short* H_BF = (unsigned short*)alloc((size_t)kB * kH * 2);
  float* CTXC_F = (float*)alloc((size_t)kB * kH * 4);
  float* WSEL_F = (float*)alloc((size_t)kTB * 8 * 4);
  float* VSN_F = (float*)alloc((size_t)kTB * kH * 4);
  unsigned short* VSN_BF = (unsigned short*)alloc((size_t)kTB * kH * 2);
  float* GATES_F = (float*)alloc((size_t)kB * 768 * 4);
  unsigned short* INTER_BF = (unsigned short*)alloc((size_t)kB * 512 * 2);
  float* CRAW_F = (float*)alloc((size_t)kB * kH * 4);
  float* HS_F = (float*)alloc((size_t)kTB * kH * 4);
  unsigned short* HS_BF = (unsigned short*)alloc((size_t)kTB * kH * 2);
  float* HID_F = (float*)alloc((size_t)kTB * kH * 4);
  unsigned short* HID_BF = (unsigned short*)alloc((size_t)kTB * kH * 2);
  unsigned* BAR = (unsigned*)alloc(256);
  if (off > ws_size) return;  // workspace too small: nothing safe to do

  float* out_ffn = (float*)d_out;
  float* out_hid = (float*)d_out + kBTH;

  hipMemsetAsync(BAR, 0, 256, stream);

  auto wconv = [&](const float* W, unsigned short* WT, int K, int N, int b) {
    size_t tot = (size_t)b * K * N;
    k_wconv<<<dim3((unsigned)((tot + 255) / 256)), dim3(256), 0, stream>>>(W, WT, K, N, b);
  };
  // Weight transpose+bf16 conversion (weights stay hot in 192MB L2 afterwards)
  wconv(v_w1, WT_VW1, 256, 256, 8);
  wconv(v_w2, WT_VW2, 256, 256, 8);
  wconv(v_wg, WT_VWG, 256, 512, 8);
  int scBase[4] = {23, 31, 39, 47};  // vs, enr, h, c
  for (int g = 0; g < 4; g++) {
    unsigned short* base = WT_SC + (size_t)g * 262144;
    wconv(F32(scBase[g] + 0), base, 256, 256, 1);            // w1T
    wconv(F32(scBase[g] + 2), base + 65536, 256, 256, 1);    // w2T
    wconv(F32(scBase[g] + 4), base + 131072, 256, 512, 1);   // wgT
  }
  wconv(wi, WT_GATES, 512, 256, 1);
  wconv(wf, WT_GATES + (size_t)256 * 512, 512, 256, 1);
  wconv(wo, WT_GATES + (size_t)512 * 512, 512, 256, 1);
  wconv(wci, WT_WCI, 512, 512, 1);
  wconv(wcp, WT_WCP, 512, 256, 1);
  wconv(gan_wg, WT_GAN, 256, 512, 1);
  wconv(fin_w1, WT_FIN_W1, 256, 256, 1);
  wconv(fin_w2, WT_FIN_W2, 256, 256, 1);
  wconv(fin_wg, WT_FIN_WG, 256, 512, 1);

  // Prescaler embedding (bf16 activations)
  {
    size_t tot = (size_t)kTB * kFH;
    k_embed<<<dim3((unsigned)((tot + 255) / 256)), dim3(256), 0, stream>>>(seq, pw, pb, EMB_BF);
  }
  // Ticker gather + the four static-context GRNs
  k_gather<<<dim3(64), dim3(256), 0, stream>>>(ticker_emb, tickers, TICK_F, TICK_BF);
  auto sc_grn = [&](int g, float* outF, unsigned short* outB) {
    unsigned short* base = WT_SC + (size_t)g * 262144;
    k_grn256<<<dim3(kB / 16), dim3(512), 0, stream>>>(
        TICK_BF, TICK_F,
        base, F32(scBase[g] + 1), (const float*)nullptr, 1,
        base + 65536, F32(scBase[g] + 3),
        base + 131072, F32(scBase[g] + 5),
        F32(scBase[g] + 6), F32(scBase[g] + 7),
        outF, outB, 0);
  };
  sc_grn(0, CTXVS_F, nullptr);        // ctx_vs
  sc_grn(1, CTXENR_F, nullptr);       // ctx_enr
  sc_grn(2, H0_F, H_BF);              // h0 (bf16 state)
  sc_grn(3, C_F, nullptr);            // c0 (f32 state)

  // Per-variable GRNs (the FLOP majority)
  k_vargrn<<<dim3(kTB / 16, kF), dim3(512), 0, stream>>>(
      seq, pw, pb, EMB_BF, WT_VW1, v_b1, WT_VW2, v_b2, WT_VWG, v_bg, v_g, v_be, VAR_BF);

  // Variable selection weights + weighted combine
  k_flat<<<dim3((kTB + 255) / 256), dim3(256), 0, stream>>>(
      EMB_BF, CTXVS_F, fl_w1, fl_b1, fl_wc, fl_w2, fl_b2, fl_wg, fl_bg,
      fl_g, fl_be, fl_rw, fl_rb, WSEL_F);
  {
    size_t tot = (size_t)kTB * kH;
    k_combine<<<dim3((unsigned)((tot + 255) / 256)), dim3(256), 0, stream>>>(
        VAR_BF, WSEL_F, VSN_F, VSN_BF);
  }

  // Sequential xLSTM scan: persistent grid of 64 WGs, grid barrier per stage
  k_xlstm<<<dim3(64), dim3(256), 0, stream>>>(
      VSN_BF, WT_GATES, bi, bff, bo, WT_WCI, bci, WT_WCP, bcp,
      lnc_g, lnc_b, lnh_g, lnh_b,
      C_F, H_BF, GATES_F, INTER_BF, CRAW_F, HS_F, HS_BF, BAR);

  // GateAddNorm -> hidden (second output, swizzled to [B,T,H])
  k_gan<<<dim3(kTB / 16), dim3(512), 0, stream>>>(
      HS_BF, VSN_F, WT_GAN, gan_bg, gan_g, gan_b, HID_F, HID_BF, out_hid);

  // Final GRN with static-enrichment context (first output)
  k_ctxmm<<<dim3(64), dim3(256), 0, stream>>>(CTXENR_F, fin_wc, CTXC_F);
  k_grn256<<<dim3(kTB / 16), dim3(512), 0, stream>>>(
      HID_BF, HID_F,
      WT_FIN_W1, fin_b1, CTXC_F, kB,
      WT_FIN_W2, fin_b2,
      WT_FIN_WG, fin_bg,
      fin_g, fin_be,
      out_ffn, (unsigned short*)nullptr, 1);
}
